// ScaledDotProductAttention_19524921328014
// MI455X (gfx1250) — matrix operand order repaired and checked by reading, not compiled, not run on hardware
//
#include <hip/hip_runtime.h>
#include <hip/hip_bf16.h>

typedef __attribute__((ext_vector_type(16))) __bf16 v16bf;
typedef __attribute__((ext_vector_type(8)))  float  v8f;
typedef __attribute__((ext_vector_type(4)))  float  vf4;

#define B_      32
#define L_      2048
#define D_      128
#define QTILE   16
#define KTILES  (L_ / 16)     // 128
#define WAVES   8
#define THREADS (WAVES * 32)  // 256

// convert 4 f32 -> 4 bf16 into dst[o..o+3]
__device__ __forceinline__ void cvt4(v16bf& dst, int o, vf4 t) {
    dst[o + 0] = (__bf16)t[0];
    dst[o + 1] = (__bf16)t[1];
    dst[o + 2] = (__bf16)t[2];
    dst[o + 3] = (__bf16)t[3];
}

__launch_bounds__(THREADS)
__global__ void sdpa_wmma_kernel(const float* __restrict__ q,
                                 const float* __restrict__ k,
                                 const float* __restrict__ v,
                                 const unsigned char* __restrict__ mask,
                                 float* __restrict__ out,    // [B,L,D]
                                 float* __restrict__ attn)   // [B,L,L]
{
    // 16 x 2048 f32 score strip: 128 KB of the WGP's 320 KB LDS
    __shared__ float sc[QTILE * L_];
    __shared__ float red[QTILE][WAVES * 2];
    __shared__ float rowmax[QTILE];
    __shared__ float rowinv[QTILE];

    const int tid  = threadIdx.x;
    const int lane = tid & 31;
    const int wave = tid >> 5;

    const int b  = blockIdx.x / (L_ / QTILE);
    const int qt = blockIdx.x % (L_ / QTILE);
    const int q0 = qt * QTILE;

    const float* qb = q + (size_t)b * L_ * D_;
    const float* kb = k + (size_t)b * L_ * D_;
    const float* vb = v + (size_t)b * L_ * D_;
    const unsigned char* mb = mask + (size_t)b * L_ * L_ + (size_t)q0 * L_;

    // ---- WMMA lane-layout helpers (16-bit A-matrix 16x32, ISA 7.12.2) ----
    const int m    = lane & 15;           // matrix row (A) / col (B,C,D)
    const int half = (lane >> 4) * 8;     // hi lanes carry K+8 within each 16-chunk
    const int koff = (lane >> 4) * 16;    // unused; B operands use the same k order as A

    // ---- build the 4 bf16 A-fragments of the Q tile (covers D=128) ------
    v16bf aq[4];
    {
        const vf4* qrow4 = (const vf4*)(qb + (size_t)(q0 + m) * D_);
        #pragma unroll
        for (int j = 0; j < 4; ++j) {
            const int p = j * 8 + (half >> 2);            // vf4 index of d0+half
            cvt4(aq[j], 0,  qrow4[p]);
            cvt4(aq[j], 4,  qrow4[p + 1]);
            cvt4(aq[j], 8,  qrow4[p + 4]);                // d0+16+half
            cvt4(aq[j], 12, qrow4[p + 5]);
        }
    }

    const float scale = 0.08838834764831845f;  // 1/sqrt(128)

    // ========== phase 1: scores = QK^T * scale, masked, -> LDS ===========
    for (int kt = wave; kt < KTILES; kt += WAVES) {
        const int n0 = kt * 16;
        const vf4* krow4 = (const vf4*)(kb + (size_t)(n0 + m) * D_);
        v8f c = {};
        #pragma unroll
        for (int j = 0; j < 4; ++j) {
            const int p = j * 8 + (half >> 2);            // same k order as A
            v16bf bk;
            cvt4(bk, 0,  krow4[p]);
            cvt4(bk, 4,  krow4[p + 1]);
            cvt4(bk, 8,  krow4[p + 4]);
            cvt4(bk, 12, krow4[p + 5]);
            c = __builtin_amdgcn_wmma_f32_16x16x32_bf16(false, aq[j], false, bk,
                                                        (short)0, c, false, false);
        }
        const int col = n0 + m;
        #pragma unroll
        for (int r = 0; r < 8; ++r) {
            const int row = r + ((lane >> 4) << 3);   // VGPR r: M=r (lo) / r+8 (hi)
            float s = c[r] * scale;
            if (mb[(size_t)row * L_ + col]) s = -__builtin_inff();
            sc[row * L_ + col] = s;
        }
    }
    __syncthreads();

    // ========== phase 2: row softmax in LDS (f32, b128 LDS traffic) ======
    {
        const int row = tid >> 4;          // 16 rows x 16 threads each
        const int seg = tid & 15;
        vf4* srow4 = (vf4*)(sc + row * L_) + seg * (L_ / 16 / 4);  // 32 vf4 each

        float mx = -__builtin_inff();
        for (int c = 0; c < 32; ++c) {
            vf4 t = srow4[c];
            mx = fmaxf(mx, fmaxf(fmaxf(t[0], t[1]), fmaxf(t[2], t[3])));
        }
        red[row][seg] = mx;
        __syncthreads();
        if (seg == 0) {
            float m2 = red[row][0];
            #pragma unroll
            for (int i = 1; i < 16; ++i) m2 = fmaxf(m2, red[row][i]);
            rowmax[row] = m2;
        }
        __syncthreads();
        const float rm = rowmax[row];
        float sum = 0.f;
        for (int c = 0; c < 32; ++c) {
            vf4 t = srow4[c];
            vf4 e;
            e[0] = __expf(t[0] - rm);
            e[1] = __expf(t[1] - rm);
            e[2] = __expf(t[2] - rm);
            e[3] = __expf(t[3] - rm);
            srow4[c] = e;
            sum += (e[0] + e[1]) + (e[2] + e[3]);
        }
        red[row][seg] = sum;
        __syncthreads();
        if (seg == 0) {
            float s2 = 0.f;
            #pragma unroll
            for (int i = 0; i < 16; ++i) s2 += red[row][i];
            rowinv[row] = 1.0f / s2;
        }
        __syncthreads();
    }

    // ========== phase 3: write normalized attn (b128, coalesced) =========
    {
        vf4* ab4 = (vf4*)(attn + (size_t)b * L_ * L_ + (size_t)q0 * L_);
        const vf4* sc4 = (const vf4*)sc;
        for (int i = tid; i < QTILE * L_ / 4; i += THREADS) {
            const int row = i >> 9;        // / (2048/4)
            ab4[i] = sc4[i] * rowinv[row];
        }
    }

    // ========== phase 4: out = P @ V (bf16 WMMA, f32 acc) ================
    {
        const float inv = rowinv[m];
        const vf4* prow4 = (const vf4*)(sc + m * L_);
        const int n0 = wave * 16;          // each wave owns one 16-col D tile
        v8f c = {};
        for (int kt = 0; kt < L_ / 32; ++kt) {
            const int p = kt * 8 + (half >> 2);
            v16bf ap;
            cvt4(ap, 0,  prow4[p]     * inv);
            cvt4(ap, 4,  prow4[p + 1] * inv);
            cvt4(ap, 8,  prow4[p + 4] * inv);             // k0+16+half
            cvt4(ap, 12, prow4[p + 5] * inv);
            // B = V block 32x16: element (kk, n) = V[kt*32+kk][n0+n]
            const float* vrow = vb + (size_t)(kt * 32 + half) * D_ + n0 + m;
            v16bf bv;
            #pragma unroll
            for (int e = 0; e < 16; ++e)
                bv[e] = (__bf16)vrow[(size_t)(e + (e & 8)) * D_];
            c = __builtin_amdgcn_wmma_f32_16x16x32_bf16(false, ap, false, bv,
                                                        (short)0, c, false, false);
        }
        float* ob = out + (size_t)b * L_ * D_ + (size_t)q0 * D_;
        #pragma unroll
        for (int r = 0; r < 8; ++r) {
            const int row = r + ((lane >> 4) << 3);
            ob[(size_t)row * D_ + n0 + m] = c[r];
        }
    }
}

extern "C" void kernel_launch(void* const* d_in, const int* in_sizes, int n_in,
                              void* d_out, int out_size, void* d_ws, size_t ws_size,
                              hipStream_t stream) {
    (void)in_sizes; (void)n_in; (void)out_size; (void)d_ws; (void)ws_size;
    const float* q = (const float*)d_in[0];
    const float* k = (const float*)d_in[1];
    const float* v = (const float*)d_in[2];
    const unsigned char* mask = (const unsigned char*)d_in[3];

    float* out  = (float*)d_out;                       // [B,L,D] first
    float* attn = out + (size_t)B_ * L_ * D_;          // then [B,L,L]

    dim3 grid(B_ * (L_ / QTILE));   // 4096 workgroups
    dim3 block(THREADS);            // 256 threads = 8 wave32
    sdpa_wmma_kernel<<<grid, block, 0, stream>>>(q, k, v, mask, out, attn);
}
